// RoPESelfAttention_18940805775985
// MI455X (gfx1250) — compile-verified, matrix operand order checked by reading, not run on hardware
//
#include <hip/hip_runtime.h>
#include <math.h>

// ---------------------------------------------------------------------------
// RoPE self-attention for MI455X (gfx1250). All matmuls via v_wmma_f32_16x16x32_f16.
// ---------------------------------------------------------------------------

typedef _Float16 half_t;
typedef __attribute__((ext_vector_type(16))) _Float16 v16h;
typedef __attribute__((ext_vector_type(8)))  _Float16 v8h;
typedef __attribute__((ext_vector_type(8)))  float    v8f;

#define BATCH  4
#define SEQ    2048
#define DMODEL 1024
#define NHEAD  16
#define DHEAD  64

// ---------------------------------------------------------------------------
// WMMA helpers.
// 16-bit A-matrix 16x32 layout (ISA 7.12.2): lane L (L&15 = row), contraction
// elements: lanes 0-15 hold K = {0..7, 16..23}, lanes 16-31 hold K = {8..15, 24..31}.
// For row-major storage with contiguous contraction dim this is two 16B loads.
// The B operand uses the same loader on the *transposed* matrix (N x K).
// ---------------------------------------------------------------------------
__device__ inline v16h load_frag16(const half_t* base, int ld, int row0, int k0, int lane) {
    const int r  = row0 + (lane & 15);
    const int kb = k0 + ((lane >> 4) << 3);
    const half_t* p = base + (size_t)r * ld + kb;
    v16h f;
    *reinterpret_cast<v8h*>(&f)       = *reinterpret_cast<const v8h*>(p);        // K = kb..kb+7
    *(reinterpret_cast<v8h*>(&f) + 1) = *reinterpret_cast<const v8h*>(p + 16);   // K = kb+16..kb+23
    return f;
}

__device__ inline v8f wmma_f16(v16h a, v16h b, v8f c) {
    return __builtin_amdgcn_wmma_f32_16x16x32_f16(false, a, false, b, (short)0, c, false, false);
}

// ---------------------------------------------------------------------------
// Kernel 1a: plain f32 -> f16 convert
// ---------------------------------------------------------------------------
__global__ void cvt_f16_kernel(const float* __restrict__ in, half_t* __restrict__ out, int n) {
    int i = blockIdx.x * blockDim.x + threadIdx.x;
    if (i < n) out[i] = (half_t)in[i];
}

// ---------------------------------------------------------------------------
// Kernel 1b: transpose-convert weight (K x N f32) -> (N x K f16)
// ---------------------------------------------------------------------------
__global__ void cvtT_f16_kernel(const float* __restrict__ in, half_t* __restrict__ out, int K, int N) {
    int i = blockIdx.x * blockDim.x + threadIdx.x;
    if (i < N * K) {
        int n = i / K, k = i % K;
        out[i] = (half_t)in[(size_t)k * N + n];
    }
}

// ---------------------------------------------------------------------------
// Kernel 2/5: generic WMMA GEMM.  C[M,N] = A[M,K] @ Bt[N,K]^T + bias.
// Block = 256 threads (8 waves); each wave owns a 16x64 tile.
// grid.x = M/16, grid.y = N/512.  Output: f32 (Cf) or f16 (Ch).
// ---------------------------------------------------------------------------
__global__ __launch_bounds__(256) void gemm_wmma_kernel(
        const half_t* __restrict__ A, const half_t* __restrict__ Bt,
        const float* __restrict__ bias,
        float* __restrict__ Cf, half_t* __restrict__ Ch,
        int M, int N, int K) {
    const int lane = threadIdx.x & 31;
    const int wave = threadIdx.x >> 5;
    const int m0 = blockIdx.x * 16;
    const int n0 = (blockIdx.y * 8 + wave) * 64;

    v8f acc[4] = {};
    for (int k = 0; k < K; k += 32) {
        v16h a = load_frag16(A, K, m0, k, lane);
#pragma unroll
        for (int t = 0; t < 4; ++t) {
            v16h b = load_frag16(Bt, K, n0 + t * 16, k, lane);
            acc[t] = wmma_f16(a, b, acc[t]);
        }
    }
    // C/D layout: lane holds col = lane&15; rows = (lane>>4)*8 + j
    const int col   = lane & 15;
    const int rbase = (lane >> 4) << 3;
#pragma unroll
    for (int t = 0; t < 4; ++t) {
        int n = n0 + t * 16 + col;
        float bv = bias ? bias[n] : 0.f;
#pragma unroll
        for (int j = 0; j < 8; ++j) {
            size_t idx = (size_t)(m0 + rbase + j) * N + n;
            float v = acc[t][j] + bv;
            if (Cf) Cf[idx] = v;
            else    Ch[idx] = (half_t)v;
        }
    }
}

// ---------------------------------------------------------------------------
// Kernel 3: RoPE + split heads.
// qkv: [B*T, 3*DMODEL] f16 -> q,k: [B,H,T,64] (RoPE'd), vT: [B,H,64,T].
// One thread per (b,h,t,d) with d in [0,32).
// ---------------------------------------------------------------------------
__global__ void rope_split_kernel(const half_t* __restrict__ qkv,
                                  half_t* __restrict__ q, half_t* __restrict__ k,
                                  half_t* __restrict__ vT) {
    int i = blockIdx.x * blockDim.x + threadIdx.x;
    int d = i & 31;
    int t = (i >> 5) & (SEQ - 1);
    int h = (i >> 16) & (NHEAD - 1);
    int b = i >> 20;
    if (b >= BATCH) return;

    size_t row = ((size_t)b * SEQ + t) * (3 * DMODEL);
    int c0 = h * DHEAD + d;
    float q1 = (float)qkv[row + c0],              q2 = (float)qkv[row + c0 + 32];
    float k1 = (float)qkv[row + DMODEL + c0],     k2 = (float)qkv[row + DMODEL + c0 + 32];
    float v1 = (float)qkv[row + 2 * DMODEL + c0], v2 = (float)qkv[row + 2 * DMODEL + c0 + 32];

    float invf = __expf(-logf(10000.f) * (float)d * (1.f / 32.f));
    float ang  = (float)t * invf;
    float sn, cs;
    __sincosf(ang, &sn, &cs);

    size_t base = (((size_t)b * NHEAD + h) * SEQ + t) * DHEAD;
    q[base + d]      = (half_t)(q1 * cs - q2 * sn);
    q[base + d + 32] = (half_t)(q1 * sn + q2 * cs);
    k[base + d]      = (half_t)(k1 * cs - k2 * sn);
    k[base + d + 32] = (half_t)(k1 * sn + k2 * cs);

    size_t vb = ((size_t)b * NHEAD + h) * DHEAD;
    vT[(vb + d) * SEQ + t]      = (half_t)v1;
    vT[(vb + d + 32) * SEQ + t] = (half_t)v2;
}

// ---------------------------------------------------------------------------
// Kernel 4: flash attention.  Block = 128 threads (4 waves); each wave owns
// 16 query rows; streams keys in chunks of 32.  S via 4 WMMAs, online softmax
// via wave32 shuffles, P staged through LDS (C-layout -> A-layout), P@V via
// 4 WMMAs into f32 accumulators.  grid.x = B*H*(SEQ/64).
// ---------------------------------------------------------------------------
__global__ __launch_bounds__(128) void attn_wmma_kernel(
        const half_t* __restrict__ Q, const half_t* __restrict__ Kc,
        const half_t* __restrict__ Vt, half_t* __restrict__ Y) {
    const int lane = threadIdx.x & 31;
    const int wave = threadIdx.x >> 5;
    const int qt   = blockIdx.x % (SEQ / 64);
    const int bh   = blockIdx.x / (SEQ / 64);
    const int q0   = qt * 64 + wave * 16;

    const half_t* Qp = Q  + (size_t)bh * SEQ * DHEAD;
    const half_t* Kp = Kc + (size_t)bh * SEQ * DHEAD;
    const half_t* Vp = Vt + (size_t)bh * DHEAD * SEQ;

    __shared__ __align__(16) half_t Pl[4][16][32];

    v16h qa0 = load_frag16(Qp, DHEAD, q0, 0, lane);
    v16h qa1 = load_frag16(Qp, DHEAD, q0, 32, lane);

    v8f o[4] = {};
    float mrow[8], lrow[8];
#pragma unroll
    for (int j = 0; j < 8; ++j) { mrow[j] = -1e30f; lrow[j] = 0.f; }
    const float scale = 0.125f;  // 1/sqrt(64)

    const int col  = lane & 15;
    const int hsel = lane >> 4;

    for (int kb = 0; kb < SEQ; kb += 32) {
        // ---- S = Q @ K^T for 32 keys (two 16x16 tiles, Dh=64 = two k-steps) ----
        v8f s0 = {}, s1 = {};
        {
            v16h b00 = load_frag16(Kp, DHEAD, kb,      0,  lane);
            v16h b01 = load_frag16(Kp, DHEAD, kb,      32, lane);
            s0 = wmma_f16(qa0, b00, s0);
            s0 = wmma_f16(qa1, b01, s0);
            v16h b10 = load_frag16(Kp, DHEAD, kb + 16, 0,  lane);
            v16h b11 = load_frag16(Kp, DHEAD, kb + 16, 32, lane);
            s1 = wmma_f16(qa0, b10, s1);
            s1 = wmma_f16(qa1, b11, s1);
        }
        // ---- online softmax (per row = hsel*8 + j; 16 lanes per half share rows) ----
#pragma unroll
        for (int j = 0; j < 8; ++j) {
            float e0 = s0[j] * scale, e1 = s1[j] * scale;
            float mx = fmaxf(e0, e1);
            for (int mm = 1; mm < 16; mm <<= 1) mx = fmaxf(mx, __shfl_xor(mx, mm, 32));
            float mnew  = fmaxf(mrow[j], mx);
            float alpha = __expf(mrow[j] - mnew);
            float p0 = __expf(e0 - mnew), p1 = __expf(e1 - mnew);
            float rs = p0 + p1;
            for (int mm = 1; mm < 16; mm <<= 1) rs += __shfl_xor(rs, mm, 32);
            lrow[j] = lrow[j] * alpha + rs;
            mrow[j] = mnew;
            o[0][j] *= alpha; o[1][j] *= alpha; o[2][j] *= alpha; o[3][j] *= alpha;
            int r = hsel * 8 + j;
            Pl[wave][r][col]      = (half_t)p0;
            Pl[wave][r][col + 16] = (half_t)p1;
        }
        __syncthreads();  // uniform trip count across all 4 waves; orders LDS

        // ---- O += P @ V  (A-frag from LDS, B-frags from V^T, contiguous) ----
        v16h pa = load_frag16(&Pl[wave][0][0], 32, 0, 0, lane);
#pragma unroll
        for (int t = 0; t < 4; ++t) {
            v16h bv = load_frag16(Vp, SEQ, t * 16, kb, lane);
            o[t] = wmma_f16(pa, bv, o[t]);
        }
        __syncthreads();
    }

    // ---- normalize and write back to [B,T,D] layout ----
    const int b = bh / NHEAD, h = bh % NHEAD;
    const int rbase = hsel * 8;
#pragma unroll
    for (int t = 0; t < 4; ++t) {
#pragma unroll
        for (int j = 0; j < 8; ++j) {
            int r = q0 + rbase + j;
            float v = o[t][j] / lrow[j];
            Y[((size_t)b * SEQ + r) * DMODEL + h * DHEAD + t * 16 + col] = (half_t)v;
        }
    }
}

// ---------------------------------------------------------------------------
// Host launch
// ---------------------------------------------------------------------------
extern "C" void kernel_launch(void* const* d_in, const int* in_sizes, int n_in,
                              void* d_out, int out_size, void* d_ws, size_t ws_size,
                              hipStream_t stream) {
    (void)in_sizes; (void)n_in; (void)out_size; (void)ws_size;

    const float* x     = (const float*)d_in[0];
    const float* Wqkv  = (const float*)d_in[1];
    const float* bqkv  = (const float*)d_in[2];
    const float* Wout  = (const float*)d_in[3];
    const float* bout  = (const float*)d_in[4];
    float*       out   = (float*)d_out;

    char* ws = (char*)d_ws;
    size_t off = 0;
    auto wsalloc = [&](size_t bytes) -> void* {
        void* p = ws + off;
        off += (bytes + 255) & ~(size_t)255;
        return p;
    };
    const size_t NTOK = (size_t)BATCH * SEQ;                     // 8192
    half_t* xh    = (half_t*)wsalloc(NTOK * DMODEL * 2);         // 16 MB (reused as yh)
    half_t* wqkvT = (half_t*)wsalloc((size_t)3 * DMODEL * DMODEL * 2);
    half_t* woutT = (half_t*)wsalloc((size_t)DMODEL * DMODEL * 2);
    half_t* qkvh  = (half_t*)wsalloc(NTOK * 3 * DMODEL * 2);     // 48 MB
    half_t* qh    = (half_t*)wsalloc(NTOK * DMODEL * 2);
    half_t* kh    = (half_t*)wsalloc(NTOK * DMODEL * 2);
    half_t* vTh   = (half_t*)wsalloc(NTOK * DMODEL * 2);
    half_t* yh    = xh;  // x is dead after the QKV GEMM; alias to save workspace

    // 1) convert activations + transpose-convert weights to f16
    {
        int n = (int)(NTOK * DMODEL);
        cvt_f16_kernel<<<(n + 255) / 256, 256, 0, stream>>>(x, xh, n);
    }
    {
        int n = 3 * DMODEL * DMODEL;
        cvtT_f16_kernel<<<(n + 255) / 256, 256, 0, stream>>>(Wqkv, wqkvT, DMODEL, 3 * DMODEL);
    }
    {
        int n = DMODEL * DMODEL;
        cvtT_f16_kernel<<<(n + 255) / 256, 256, 0, stream>>>(Wout, woutT, DMODEL, DMODEL);
    }

    // 2) QKV GEMM: [8192,1024] @ [1024,3072] -> f16
    {
        dim3 grid((unsigned)(NTOK / 16), (3 * DMODEL) / 512);
        gemm_wmma_kernel<<<grid, 256, 0, stream>>>(xh, wqkvT, bqkv,
                                                   nullptr, qkvh,
                                                   (int)NTOK, 3 * DMODEL, DMODEL);
    }

    // 3) RoPE + head split (+ V transpose)
    {
        int n = BATCH * NHEAD * SEQ * 32;
        rope_split_kernel<<<n / 256, 256, 0, stream>>>(qkvh, qh, kh, vTh);
    }

    // 4) flash attention
    {
        int blocks = BATCH * NHEAD * (SEQ / 64);
        attn_wmma_kernel<<<blocks, 128, 0, stream>>>(qh, kh, vTh, yh);
    }

    // 5) output GEMM with f32 store: [8192,1024] @ [1024,1024] + b_out
    {
        dim3 grid((unsigned)(NTOK / 16), DMODEL / 512);
        gemm_wmma_kernel<<<grid, 256, 0, stream>>>(yh, woutT, bout,
                                                   out, nullptr,
                                                   (int)NTOK, DMODEL, DMODEL);
    }
}
